// InteractLayer_3307124818154
// MI455X (gfx1250) — matrix operand order repaired and checked by reading, not compiled, not run on hardware
//
#include <hip/hip_runtime.h>

// ---------------------------------------------------------------------------
// InteractLayer (HIP-NN style) for gfx1250 / MI455X, wave32 + WMMA bf16.
//   out[a,:] = (env[a,:] | x[a,:]) @ (Wint_rs | Wselfᵀ) + b_self
// env accumulated in fp32 registers -> bf16 LDS tile -> v_wmma_f32_16x16x32_bf16.
// ---------------------------------------------------------------------------

#define N_ATOMS   30000
#define NF        64
#define ND        20
#define K_ENV     (ND * NF)        // 1280
#define K_TOT     (K_ENV + NF)     // 1344 (self-interaction fused as extra K)
#define NCHUNK    (K_TOT / 32)     // 42 WMMA k-steps
#define TILE_M    16
#define HARD_CUT  6.5f
#define COS_K     0.24166097335f   // 0.5*pi/HARD_CUT

typedef __attribute__((ext_vector_type(16))) __bf16   v16bf;
typedef __attribute__((ext_vector_type(8)))  float    v8f;
typedef __attribute__((ext_vector_type(4)))  unsigned u32x4;
typedef __attribute__((ext_vector_type(2)))  float    f32x2;

union Frag32B { u32x4 q[2]; v16bf v; };        // 32B = 16 bf16 = one WMMA operand/lane
union Pack2   { __bf16 h[2]; unsigned u; };    // two bf16 packed into one dword

// ---------------------------- CSR build ------------------------------------

__global__ void zero_i32(int* __restrict__ p, int n) {
    int i = blockIdx.x * 256 + threadIdx.x;
    if (i < n) p[i] = 0;
}

__global__ void count_k(const int* __restrict__ pf, int* __restrict__ counts, int P) {
    int i = blockIdx.x * 256 + threadIdx.x;
    if (i < P) atomicAdd(&counts[pf[i]], 1);
}

// single-block exclusive scan over 30000 counts (1024 thr x 30 items)
__global__ void scan_k(const int* __restrict__ counts, int* __restrict__ offs,
                       int* __restrict__ cursor, int P) {
    __shared__ int sums[1024];
    const int CH = (N_ATOMS + 1023) / 1024;   // 30
    int t = threadIdx.x;
    int base = t * CH;
    int vals[CH];
    int local = 0;
    #pragma unroll
    for (int i = 0; i < CH; ++i) {
        int idx = base + i;
        vals[i] = (idx < N_ATOMS) ? counts[idx] : 0;
        local += vals[i];
    }
    sums[t] = local;
    __syncthreads();
    // Hillis-Steele inclusive scan over 1024 partials
    for (int off = 1; off < 1024; off <<= 1) {
        int v = (t >= off) ? sums[t - off] : 0;
        __syncthreads();
        sums[t] += v;
        __syncthreads();
    }
    int run = (t > 0) ? sums[t - 1] : 0;
    #pragma unroll
    for (int i = 0; i < CH; ++i) {
        int idx = base + i;
        if (idx < N_ATOMS) {
            offs[idx] = run;
            cursor[idx] = run;
            run += vals[i];
        }
    }
    if (t == 1023) offs[N_ATOMS] = P;
}

__global__ void scatter_k(const int* __restrict__ pf, int* __restrict__ cursor,
                          int* __restrict__ perm, int P) {
    int i = blockIdx.x * 256 + threadIdx.x;
    if (i < P) {
        int a = pf[i];
        int pos = atomicAdd(&cursor[a], 1);
        perm[pos] = i;
    }
}

// --------------------- weight pre-pack into WMMA-B layout -------------------
// wprep[chunk][ntile][lane][i] bf16: lane's 16 B-matrix values contiguous.
// B layout (bf16 32x16): lanes 0-15 hold K = k0..k0+15, lanes 16-31 K = k0+16..k0+31,
// column N = lane & 15 (contiguous K order, same as the A fragment read from LDS).

__global__ void wprep_kernel(const float* __restrict__ int_w,
                             const float* __restrict__ w_self,
                             __bf16* __restrict__ wprep) {
    int idx = blockIdx.x * 256 + threadIdx.x;
    if (idx >= NCHUNK * 4 * 32 * 16) return;
    int i    = idx & 15;
    int lane = (idx >> 4) & 31;
    int t    = (idx >> 9) & 3;
    int c    = idx >> 11;
    int half = lane >> 4;
    int n    = t * 16 + (lane & 15);
    int K    = c * 32 + half * 16 + i;
    float v;
    if (K < K_ENV) {
        // weights_rs[K][n] = int_weights[nu][n][fin], K = nu*64 + fin
        int nu = K >> 6, fin = K & 63;
        v = int_w[(nu * NF + n) * NF + fin];
    } else {
        // self block: w_selfT[k2][n] = w_self[n][k2]
        int k2 = K - K_ENV;
        v = w_self[n * NF + k2];
    }
    wprep[idx] = (__bf16)v;
}

// ------------------------------- main kernel --------------------------------

__launch_bounds__(128)
__global__ void interact_kernel(const float* __restrict__ xf,
                                const int*   __restrict__ psecond,
                                const float* __restrict__ dist,
                                const float* __restrict__ mu,
                                const float* __restrict__ sigma,
                                const float* __restrict__ b_self,
                                const int*   __restrict__ offs,
                                const int*   __restrict__ perm,
                                const __bf16* __restrict__ wprep,
                                float* __restrict__ out) {
    __shared__ __bf16 envs[TILE_M][K_TOT];     // 16 x 1344 bf16 = 42 KB LDS

    const int a0   = blockIdx.x * TILE_M;
    const int tid  = threadIdx.x;
    const int wave = tid >> 5;
    const int lane = tid & 31;

    // stage self-features into K columns 1280..1343 (bf16)
    {
        int m  = tid >> 3;            // 0..15
        int c0 = (tid & 7) << 3;      // 0..56
        const float* src = xf + (a0 + m) * NF + c0;
        #pragma unroll
        for (int i = 0; i < 8; i += 2) {
            f32x2 v = *(const f32x2*)(src + i);
            Pack2 pk; pk.h[0] = (__bf16)v.x; pk.h[1] = (__bf16)v.y;
            *(unsigned*)&envs[m][K_ENV + c0 + i] = pk.u;
        }
    }

    float mus[ND], isg[ND];
    #pragma unroll
    for (int nu = 0; nu < ND; ++nu) { mus[nu] = mu[nu]; isg[nu] = 1.0f / sigma[nu]; }

    // Phase 1: envsum. Each wave owns 4 atoms; lane owns features 2*lane, 2*lane+1.
    // Accumulate fp32 in registers across the atom's pair list, then one LDS write.
    for (int j = 0; j < 4; ++j) {
        int m = wave * 4 + j;
        int a = a0 + m;
        float acc[2 * ND];
        #pragma unroll
        for (int k = 0; k < 2 * ND; ++k) acc[k] = 0.0f;

        int s = offs[a], e = offs[a + 1];
        for (int q = s; q < e; ++q) {
            int   p  = perm[q];
            float d  = dist[p];
            int   nb = psecond[p];
            f32x2 v  = *(const f32x2*)(xf + nb * NF + 2 * lane);
            float invd = 1.0f / d;
            float cc   = __cosf(COS_K * d);
            float cut  = (d < HARD_CUT) ? cc * cc : 0.0f;
            #pragma unroll
            for (int nu = 0; nu < ND; ++nu) {
                float t  = (invd - mus[nu]) * isg[nu];
                float sv = __expf(-0.5f * t * t) * cut;
                acc[2 * nu]     = fmaf(sv, v.x, acc[2 * nu]);
                acc[2 * nu + 1] = fmaf(sv, v.y, acc[2 * nu + 1]);
            }
        }
        #pragma unroll
        for (int nu = 0; nu < ND; ++nu) {
            Pack2 pk;
            pk.h[0] = (__bf16)acc[2 * nu];
            pk.h[1] = (__bf16)acc[2 * nu + 1];
            *(unsigned*)&envs[m][nu * NF + 2 * lane] = pk.u;
        }
    }
    __syncthreads();

    // Phase 2: [16 x 1344] @ [1344 x 64] via 42 x v_wmma_f32_16x16x32_bf16.
    // A-frag (matches wprep B order): lane<16 -> M=lane, K=0..15+k0;
    // lane>=16 -> M=lane-16, K=16..31+k0 -> two contiguous 16B ds loads.
    const int n0   = wave * 16;
    const int half = lane >> 4;
    const int arow = lane & 15;
    const int ncol = lane & 15;

    v8f c;
    #pragma unroll
    for (int i = 0; i < 8; ++i) c[i] = 0.0f;

    const u32x4* wp = (const u32x4*)wprep;
    for (int ck = 0; ck < NCHUNK; ++ck) {
        Frag32B af, bfr;
        const int kb = ck * 32 + half * 16;
        af.q[0] = *(const u32x4*)&envs[arow][kb];
        af.q[1] = *(const u32x4*)&envs[arow][kb + 8];
        const u32x4* wpc = wp + ((((ck * 4 + wave) * 32) + lane) << 1);
        bfr.q[0] = wpc[0];
        bfr.q[1] = wpc[1];
        c = __builtin_amdgcn_wmma_f32_16x16x32_bf16(false, af.v, false, bfr.v,
                                                    (short)0, c, false, false);
    }

    // C/D layout: VGPR v -> row (v + half*8), col = lane&15. Add bias, store.
    float bias = b_self[n0 + ncol];
    #pragma unroll
    for (int v = 0; v < 8; ++v) {
        out[(a0 + v + half * 8) * NF + n0 + ncol] = c[v] + bias;
    }
}

// ------------------------------- launcher -----------------------------------

extern "C" void kernel_launch(void* const* d_in, const int* in_sizes, int n_in,
                              void* d_out, int out_size, void* d_ws, size_t ws_size,
                              hipStream_t stream) {
    const float* in_features = (const float*)d_in[0];
    const int*   pair_first  = (const int*)  d_in[1];
    const int*   pair_second = (const int*)  d_in[2];
    const float* dist_pairs  = (const float*)d_in[3];
    const float* mu          = (const float*)d_in[4];
    const float* sigma       = (const float*)d_in[5];
    const float* int_w       = (const float*)d_in[6];
    const float* w_self      = (const float*)d_in[7];
    const float* b_self      = (const float*)d_in[8];
    float* out = (float*)d_out;
    const int P = in_sizes[1];

    // workspace layout (~1.7 MB): counts | offs | cursor | perm | wprep
    int* counts = (int*)d_ws;
    int* offs   = counts + 32768;
    int* cursor = offs   + 32768;
    int* perm   = cursor + 32768;
    __bf16* wprep = (__bf16*)(perm + 300032);

    zero_i32<<<(N_ATOMS + 255) / 256, 256, 0, stream>>>(counts, N_ATOMS);
    count_k<<<(P + 255) / 256, 256, 0, stream>>>(pair_first, counts, P);
    scan_k<<<1, 1024, 0, stream>>>(counts, offs, cursor, P);
    scatter_k<<<(P + 255) / 256, 256, 0, stream>>>(pair_first, cursor, perm, P);
    wprep_kernel<<<(NCHUNK * 4 * 32 * 16 + 255) / 256, 256, 0, stream>>>(int_w, w_self, wprep);
    interact_kernel<<<N_ATOMS / TILE_M, 128, 0, stream>>>(
        in_features, pair_second, dist_pairs, mu, sigma, b_self,
        offs, perm, wprep, out);
}
